// GraphReasoning_83073257439330
// MI455X (gfx1250) — matrix operand order repaired and checked by reading, not compiled, not run on hardware
//
#include <hip/hip_runtime.h>
#include <hip/hip_bf16.h>

typedef unsigned short u16;
typedef unsigned int   u32;
typedef __attribute__((ext_vector_type(16))) __bf16         v16bf;
typedef __attribute__((ext_vector_type(8)))  float          v8f;
typedef __attribute__((ext_vector_type(4)))  float          v4f;
typedef __attribute__((ext_vector_type(16))) unsigned short v16u;
typedef __attribute__((ext_vector_type(8)))  unsigned short v8u;
typedef __attribute__((ext_vector_type(4)))  unsigned short v4u;

#define B_ 4096
#define D_ 512
#define N_ 65536
#define TOPK 5
#define TEMP_ 10.0f

// ---- fp32 -> bf16 round-to-nearest-even ----
__device__ __forceinline__ u16 f2bf(float f) {
    unsigned int u = __float_as_uint(f);
    u += 0x7fffu + ((u >> 16) & 1u);
    return (u16)(u >> 16);
}

__device__ __forceinline__ v16bf as_bf16(v16u u) {
    return __builtin_bit_cast(v16bf, u);
}

// Branchless insertion into sorted-descending top-5 (v_cndmask chains, no
// control flow). Tie-break: lower index wins (matches jax.lax.top_k
// stability). A candidate worse than all entries is naturally a no-op.
__device__ __forceinline__ void topk_insert(float (&tv)[TOPK], int (&ti)[TOPK],
                                            float cv, int ci) {
#pragma unroll
    for (int j = 0; j < TOPK; ++j) {
        bool better = (cv > tv[j]) || ((cv == tv[j]) && (ci < ti[j]));
        float nv = better ? cv : tv[j];
        int   ni = better ? ci : ti[j];
        float ov = better ? tv[j] : cv;
        int   oi = better ? ti[j] : ci;
        tv[j] = nv; ti[j] = ni;
        cv = ov; ci = oi;
    }
}

// ---------------------------------------------------------------------------
// Kernel 1: per-row L2 normalize + bf16 convert (optionally also raw bf16)
// one block per row of 512
// ---------------------------------------------------------------------------
__global__ __launch_bounds__(256) void k_rownorm_bf16(
    const float* __restrict__ src, u16* __restrict__ dstn,
    u16* __restrict__ dstraw) {
    __shared__ float red[256];
    const int tid = threadIdx.x;
    const size_t row = blockIdx.x;
    const float2* s2 = reinterpret_cast<const float2*>(src + row * D_);
    float2 x = s2[tid];
    red[tid] = x.x * x.x + x.y * x.y;
    __syncthreads();
    for (int st = 128; st > 0; st >>= 1) {
        if (tid < st) red[tid] += red[tid + st];
        __syncthreads();
    }
    float rn = 1.0f / fmaxf(sqrtf(red[0]), 1e-12f);
    u32* dn = reinterpret_cast<u32*>(dstn + row * D_);
    dn[tid] = (u32)f2bf(x.x * rn) | ((u32)f2bf(x.y * rn) << 16);
    if (dstraw) {
        u32* dr = reinterpret_cast<u32*>(dstraw + row * D_);
        dr[tid] = (u32)f2bf(x.x) | ((u32)f2bf(x.y) << 16);
    }
}

// ---------------------------------------------------------------------------
// Kernel 2: elementwise fp32 -> bf16 (weights)
// ---------------------------------------------------------------------------
__global__ __launch_bounds__(256) void k_convert_bf16(
    const float* __restrict__ src, u16* __restrict__ dst, int n) {
    int i = blockIdx.x * 256 + threadIdx.x;
    if (i < n) dst[i] = f2bf(src[i]);
}

// ---------------------------------------------------------------------------
// Kernel 3: fused  sim GEMM (bf16 WMMA, fp32 acc) + streaming top-5
//           + softmax + weighted neighbor gather (raw fp32 memory)
// grid = B/16 blocks, 256 threads (8 waves). Each wave streams 64 cols/iter.
// ---------------------------------------------------------------------------
__global__ __launch_bounds__(256) void k_sim_topk_weighted(
    const u16* __restrict__ qbf,     // [B,D] normalized bf16 queries
    const u16* __restrict__ mbf,     // [N,D] normalized bf16 memory
    const float* __restrict__ memory,// [N,D] raw fp32 memory (for gather)
    u16* __restrict__ wbf) {         // [B,D] weighted neighbors, bf16 out
    // A fragments pre-permuted: one contiguous 32B chunk per (kfrag, lane)
    __shared__ __align__(64) u16 Aperm[16][32][16];     // 16 KB
    __shared__ __align__(64) float simbuf[8][16][64];   // 32 KB (per-wave)
    __shared__ float mval[16][16][TOPK];                // 5 KB
    __shared__ int   midx[16][16][TOPK];                // 5 KB
    __shared__ float aff[16][TOPK];
    __shared__ int   nidl[16][TOPK];

    const int tid  = threadIdx.x;
    const int w    = tid >> 5;
    const int l    = tid & 31;
    const int ln16 = l & 15;
    const int hi   = l >> 4;
    const int rbase = blockIdx.x * 16;

    // ---- fill Aperm: element e of frag = K = kblk + hi*16 + e
    for (int s = tid; s < 512; s += 256) {
        int kf = s >> 5, sl = s & 31;
        int m = sl & 15, shi = sl >> 4;
        int kblk = kf * 32;
        const u16* qrow = qbf + (size_t)(rbase + m) * D_;
#pragma unroll
        for (int i = 0; i < 16; ++i) {
            int K = kblk + shi * 16 + i;
            Aperm[kf][sl][i] = qrow[K];
        }
    }
    __syncthreads();

    float tv[TOPK]; int ti[TOPK];
#pragma unroll
    for (int j = 0; j < TOPK; ++j) { tv[j] = -3.4e38f; ti[j] = 0x7fffffff; }

    for (int it = 0; it < N_ / 512; ++it) {
        const int nbase = it * 512 + w * 64;
#pragma unroll 1
        for (int t = 0; t < 4; ++t) {
            const int ncol = nbase + t * 16 + ln16;         // memory row = sim col
            const u16* bptr = mbf + (size_t)ncol * D_ + hi * 16;
            if (it + 1 < N_ / 512)
                __builtin_prefetch(mbf + (size_t)(ncol + 512) * D_, 0, 1);
            v8f acc = {};
#pragma unroll 4
            for (int kf = 0; kf < 16; ++kf) {
                v16bf a = *reinterpret_cast<const v16bf*>(&Aperm[kf][l][0]);
                v16u bu = *reinterpret_cast<const v16u*>(bptr + kf * 32);
                acc = __builtin_amdgcn_wmma_f32_16x16x32_bf16(
                    false, a, false, as_bf16(bu), (short)0, acc, false, false);
            }
#pragma unroll
            for (int v = 0; v < 8; ++v)
                simbuf[w][v + 8 * hi][t * 16 + ln16] = acc[v];
        }
        __syncthreads();
        // ---- streaming top-5 scan: 2 lanes per row, 32 candidates each as
        //      8x float4 LDS loads. Within a lane the candidate index is
        //      strictly increasing, so an equal-to-5th-best value can never
        //      displace (stability keeps the earlier index) -- a strict
        //      max(sv) > tv[4] test is an EXACT whole-vector reject.
        {
            const int r  = l >> 1;
            const int c0 = (l & 1) * 32;
            const v4f* srow4 =
                reinterpret_cast<const v4f*>(&simbuf[w][r][c0]);
#pragma unroll 1
            for (int q = 0; q < 8; ++q) {
                v4f sv = srow4[q];
                float mx = fmaxf(fmaxf(sv[0], sv[1]), fmaxf(sv[2], sv[3]));
                if (mx > tv[TOPK - 1]) {         // rare slow path
                    int base = nbase + c0 + q * 4;
#pragma unroll
                    for (int e = 0; e < 4; ++e)
                        topk_insert(tv, ti, sv[e], base + e);
                }
            }
        }
        __syncthreads();
    }

    // ---- dump 16 lane-local lists per row, merge with same comparator
    {
        const int r = l >> 1;
        const int listid = w * 2 + (l & 1);
#pragma unroll
        for (int j = 0; j < TOPK; ++j) {
            mval[r][listid][j] = tv[j];
            midx[r][listid][j] = ti[j];
        }
    }
    __syncthreads();

    if (tid < 16) {
        float fv[TOPK]; int fi[TOPK];
#pragma unroll
        for (int j = 0; j < TOPK; ++j) { fv[j] = -3.4e38f; fi[j] = 0x7fffffff; }
#pragma unroll 1
        for (int lt = 0; lt < 16; ++lt)
#pragma unroll 1
            for (int j = 0; j < TOPK; ++j)
                topk_insert(fv, fi, mval[tid][lt][j], midx[tid][lt][j]);
        // softmax(topk * TEMP); fv[0] is the max (sorted desc)
        float e[TOPK], ssum = 0.f;
#pragma unroll
        for (int k = 0; k < TOPK; ++k) {
            e[k] = __expf((fv[k] - fv[0]) * TEMP_);
            ssum += e[k];
        }
        float rs = 1.0f / ssum;
#pragma unroll
        for (int k = 0; k < TOPK; ++k) {
            aff[tid][k]  = e[k] * rs;
            nidl[tid][k] = fi[k];
        }
    }
    __syncthreads();

    // ---- weighted neighbor sum from RAW fp32 memory; 16 threads per row,
    //      float4 gathers + packed ushort4 bf16 stores
    {
        const int r = tid >> 4;
        const int c = tid & 15;
        float a0 = aff[r][0], a1 = aff[r][1], a2 = aff[r][2],
              a3 = aff[r][3], a4 = aff[r][4];
        const v4f* m0 = reinterpret_cast<const v4f*>(memory + (size_t)nidl[r][0] * D_);
        const v4f* m1 = reinterpret_cast<const v4f*>(memory + (size_t)nidl[r][1] * D_);
        const v4f* m2 = reinterpret_cast<const v4f*>(memory + (size_t)nidl[r][2] * D_);
        const v4f* m3 = reinterpret_cast<const v4f*>(memory + (size_t)nidl[r][3] * D_);
        const v4f* m4 = reinterpret_cast<const v4f*>(memory + (size_t)nidl[r][4] * D_);
        v4u* orow = reinterpret_cast<v4u*>(wbf + (size_t)(rbase + r) * D_);
#pragma unroll 2
        for (int dd = 0; dd < 8; ++dd) {
            int q = c + dd * 16;   // consecutive threads -> consecutive float4
            v4f s = a0 * m0[q] + a1 * m1[q] + a2 * m2[q] +
                    a3 * m3[q] + a4 * m4[q];
            v4u o = { f2bf(s[0]), f2bf(s[1]), f2bf(s[2]), f2bf(s[3]) };
            orow[q] = o;
        }
    }
}

// ---------------------------------------------------------------------------
// A-fragment assembly straight from a row-major bf16 [rows, 512] global array
// ---------------------------------------------------------------------------
__device__ __forceinline__ v16bf load_afrag(const u16* __restrict__ arow,
                                            int kblk, int hi) {
    v8u lo  = *reinterpret_cast<const v8u*>(arow + kblk + hi * 16);
    v8u hi8 = *reinterpret_cast<const v8u*>(arow + kblk + hi * 16 + 8);
    v16u au;
#pragma unroll
    for (int i = 0; i < 8; ++i) { au[i] = lo[i]; au[i + 8] = hi8[i]; }
    return __builtin_bit_cast(v16bf, au);
}

// ---------------------------------------------------------------------------
// Kernel 4: gcn_out = relu(weighted @ gcn_w^T + gcn_b), bf16 out
// 8192 (16x16) tiles, 8 per block -> 1024 blocks
// ---------------------------------------------------------------------------
__global__ __launch_bounds__(256) void k_gcn(
    const u16* __restrict__ wbf, const u16* __restrict__ gwbf,
    const float* __restrict__ gb, u16* __restrict__ gbf) {
    const int tid = threadIdx.x, w = tid >> 5, l = tid & 31;
    const int ln16 = l & 15, hi = l >> 4;
    const int id = blockIdx.x * 8 + w;
    const int tm = id >> 5, tj = id & 31;   // D_/16 = 32 col tiles
    const u16* arow = wbf  + (size_t)(tm * 16 + ln16) * D_;
    const u16* brow = gwbf + (size_t)(tj * 16 + ln16) * D_ + hi * 16;
    v8f acc = {};
#pragma unroll 4
    for (int kf = 0; kf < 16; ++kf) {
        int kblk = kf * 32;
        v16bf a = load_afrag(arow, kblk, hi);
        v16u bu = *reinterpret_cast<const v16u*>(brow + kblk);
        acc = __builtin_amdgcn_wmma_f32_16x16x32_bf16(
            false, a, false, as_bf16(bu), (short)0, acc, false, false);
    }
    const int col = tj * 16 + ln16;
    const float bias = gb[col];
#pragma unroll
    for (int v = 0; v < 8; ++v) {
        int row = tm * 16 + v + 8 * hi;
        gbf[(size_t)row * D_ + col] = f2bf(fmaxf(acc[v] + bias, 0.0f));
    }
}

// ---------------------------------------------------------------------------
// Kernel 5: out = [part, gcn_out] @ fusion_w^T + fusion_b + part   (fp32 out)
// ---------------------------------------------------------------------------
__global__ __launch_bounds__(256) void k_fusion(
    const u16* __restrict__ pbf, const u16* __restrict__ gbf,
    const u16* __restrict__ fwbf, const float* __restrict__ fb,
    const float* __restrict__ part, float* __restrict__ out) {
    const int tid = threadIdx.x, w = tid >> 5, l = tid & 31;
    const int ln16 = l & 15, hi = l >> 4;
    const int id = blockIdx.x * 8 + w;
    const int tm = id >> 5, tj = id & 31;
    const u16* a1row = pbf + (size_t)(tm * 16 + ln16) * D_;
    const u16* a2row = gbf + (size_t)(tm * 16 + ln16) * D_;
    const u16* brow  = fwbf + (size_t)(tj * 16 + ln16) * (2 * D_) + hi * 16;
    v8f acc = {};
#pragma unroll 4
    for (int kf = 0; kf < 16; ++kf) {           // first half: part @ Wp^T
        int kblk = kf * 32;
        v16bf a = load_afrag(a1row, kblk, hi);
        v16u bu = *reinterpret_cast<const v16u*>(brow + kblk);
        acc = __builtin_amdgcn_wmma_f32_16x16x32_bf16(
            false, a, false, as_bf16(bu), (short)0, acc, false, false);
    }
#pragma unroll 4
    for (int kf = 0; kf < 16; ++kf) {           // second half: gcn @ Wg^T
        int kblk = kf * 32;
        v16bf a = load_afrag(a2row, kblk, hi);
        v16u bu = *reinterpret_cast<const v16u*>(brow + D_ + kblk);
        acc = __builtin_amdgcn_wmma_f32_16x16x32_bf16(
            false, a, false, as_bf16(bu), (short)0, acc, false, false);
    }
    const int col = tj * 16 + ln16;
    const float bias = fb[col];
#pragma unroll
    for (int v = 0; v < 8; ++v) {
        size_t idx = (size_t)(tm * 16 + v + 8 * hi) * D_ + col;
        out[idx] = acc[v] + bias + part[idx];   // includes residual
    }
}

// ---------------------------------------------------------------------------
extern "C" void kernel_launch(void* const* d_in, const int* in_sizes, int n_in,
                              void* d_out, int out_size, void* d_ws, size_t ws_size,
                              hipStream_t stream) {
    const float* part     = (const float*)d_in[0];
    const float* memory   = (const float*)d_in[1];
    const float* gcn_w    = (const float*)d_in[2];
    const float* gcn_b    = (const float*)d_in[3];
    const float* fusion_w = (const float*)d_in[4];
    const float* fusion_b = (const float*)d_in[5];
    float* out = (float*)d_out;

    char* ws = (char*)d_ws;
    size_t o = 0;
    auto take = [&](size_t bytes) {
        char* p = ws + o;
        o += (bytes + 255) & ~(size_t)255;
        return p;
    };
    u16* qbf  = (u16*)take((size_t)B_ * D_ * 2);      // normalized queries
    u16* pbf  = (u16*)take((size_t)B_ * D_ * 2);      // raw queries (fusion in)
    u16* mbf  = (u16*)take((size_t)N_ * D_ * 2);      // normalized memory
    u16* gwbf = (u16*)take((size_t)D_ * D_ * 2);      // gcn_w bf16
    u16* fwbf = (u16*)take((size_t)D_ * 2 * D_ * 2);  // fusion_w bf16
    u16* wbf  = (u16*)take((size_t)B_ * D_ * 2);      // weighted neighbors
    u16* gbf  = (u16*)take((size_t)B_ * D_ * 2);      // gcn output

    k_rownorm_bf16<<<B_, 256, 0, stream>>>(part, qbf, pbf);
    k_rownorm_bf16<<<N_, 256, 0, stream>>>(memory, mbf, nullptr);
    k_convert_bf16<<<(D_ * D_ + 255) / 256, 256, 0, stream>>>(gcn_w, gwbf, D_ * D_);
    k_convert_bf16<<<(D_ * 2 * D_ + 255) / 256, 256, 0, stream>>>(fusion_w, fwbf,
                                                                  D_ * 2 * D_);
    k_sim_topk_weighted<<<B_ / 16, 256, 0, stream>>>(qbf, mbf, memory, wbf);
    k_gcn<<<(B_ / 16) * (D_ / 16) / 8, 256, 0, stream>>>(wbf, gwbf, gcn_b, gbf);
    k_fusion<<<(B_ / 16) * (D_ / 16) / 8, 256, 0, stream>>>(pbf, gbf, fwbf,
                                                            fusion_b, part, out);
}
